// _DummySwiGLUExperts_44083544326931
// MI455X (gfx1250) — matrix operand order repaired and checked by reading, not compiled, not run on hardware
//
#include <hip/hip_runtime.h>
#include <hip/hip_bf16.h>
#include <math.h>

// ---------------- problem constants (match reference) ----------------
#define NUM_EXPERTS 8
#define HID 2048          // hidden size H
#define INTER 2048        // intermediate size I
#define NUM_TOKENS 4096
#define TOPK 2
#define TK (NUM_TOKENS * TOPK)   // 8192 expanded rows

#define LDSTR 72          // padded LDS row stride in halves (144B, bank-conflict-free)

typedef __attribute__((ext_vector_type(16))) _Float16 v16h;
typedef __attribute__((ext_vector_type(8)))  float    v8f;
typedef __attribute__((ext_vector_type(2)))  __fp16   v2fp;   // cvt_pkrtz result type

union HFrag {
    v16h v;
    _Float16 h[16];
    uint4 q[2];
};

// ---------------- routing ----------------
__global__ void moe_route_zero(int* counts, int* cursors) {
    if (threadIdx.x < NUM_EXPERTS) {
        counts[threadIdx.x] = 0;
        cursors[threadIdx.x] = 0;
    }
}

__global__ void moe_route_count(const int* __restrict__ topk_idx, int* counts) {
    int i = blockIdx.x * blockDim.x + threadIdx.x;
    if (i < TK) atomicAdd(&counts[topk_idx[i]], 1);
}

__global__ void moe_route_scan(const int* __restrict__ counts, int* offsets) {
    if (threadIdx.x == 0) {
        int s = 0;
        for (int e = 0; e < NUM_EXPERTS; ++e) { offsets[e] = s; s += counts[e]; }
    }
}

__global__ void moe_route_fill(const int* __restrict__ topk_idx,
                               const float* __restrict__ topk_wgt,
                               const int* __restrict__ offsets, int* cursors,
                               int* rowTok, float* rowWgt) {
    int i = blockIdx.x * blockDim.x + threadIdx.x;
    if (i < TK) {
        int e = topk_idx[i];
        int pos = offsets[e] + atomicAdd(&cursors[e], 1);
        rowTok[pos] = i / TOPK;          // source token
        rowWgt[pos] = topk_wgt[i];       // routing weight
    }
}

// ---------------- fp32 -> fp16 hidden conversion ----------------
__global__ void moe_cvt_hs(const float* __restrict__ hs, _Float16* __restrict__ out) {
    int i = 4 * (blockIdx.x * blockDim.x + threadIdx.x);
    if (i < NUM_TOKENS * HID) {
        float4 f = *(const float4*)(hs + i);
        union { v2fp h2[2]; uint2 u; } p;
        p.h2[0] = __builtin_amdgcn_cvt_pkrtz(f.x, f.y);   // v_cvt_pk_rtz_f16_f32
        p.h2[1] = __builtin_amdgcn_cvt_pkrtz(f.z, f.w);
        *(uint2*)(out + i) = p.u;
    }
}

__global__ void moe_zero_out(float* __restrict__ out) {
    int i = 4 * (blockIdx.x * blockDim.x + threadIdx.x);
    if (i < NUM_TOKENS * HID) {
        *(float4*)(out + i) = make_float4(0.f, 0.f, 0.f, 0.f);
    }
}

// pack two float4 -> 8 f16 in a uint4 (4x v_cvt_pk_rtz_f16_f32, no fixup moves)
__device__ __forceinline__ uint4 pack8(float4 f0, float4 f1) {
    union { v2fp h2[4]; uint4 q; } p;
    p.h2[0] = __builtin_amdgcn_cvt_pkrtz(f0.x, f0.y);
    p.h2[1] = __builtin_amdgcn_cvt_pkrtz(f0.z, f0.w);
    p.h2[2] = __builtin_amdgcn_cvt_pkrtz(f1.x, f1.y);
    p.h2[3] = __builtin_amdgcn_cvt_pkrtz(f1.z, f1.w);
    return p.q;
}

// ---------------- phase A: x @ gate_up[e]^T, SwiGLU -> act (f16) ----------------
// grid: (INTER/64, TK/128, NUM_EXPERTS), block 256 = 8 waves.
// Block tile: 128 expert rows x 64 act cols (=> 128 weight rows: 64 gate + 64 up).
// Wave = (rowGroup 0..3) x (colGroup 0..1): 32 rows x 32 act cols (gate+up).
// Double-buffered pipeline: A tile stored into the free buffer at iteration top
// (only its loads are waited on); B fp32 loads stay in flight across the WMMAs.
__global__ void __launch_bounds__(256, 2)
moe_gateup_swiglu(const _Float16* __restrict__ hsH,     // [T, H] f16
                  const float* __restrict__ gup,        // [E, 2I, H] f32
                  const int* __restrict__ counts,
                  const int* __restrict__ offsets,
                  const int* __restrict__ rowTok,
                  _Float16* __restrict__ act)           // [TK, I] f16, sorted rows
{
    __shared__ __align__(16) _Float16 ldsA[2][128 * LDSTR];   // 2 x 18 KB
    __shared__ __align__(16) _Float16 ldsB[2][128 * LDSTR];   // 2 x 18 KB

    const int e = blockIdx.z;
    const int cnt = counts[e];
    const int rowStart = blockIdx.y * 128;
    if (rowStart >= cnt) return;                         // uniform early exit
    const int off = offsets[e];
    const int c0 = blockIdx.x * 64;                      // act column base

    const int tid = threadIdx.x;
    const int lane = tid & 31;
    const int wave = tid >> 5;
    const int rowGroup = wave >> 1;                      // 0..3
    const int colGroup = wave & 1;                       // 0..1
    const int laneM = lane & 15;
    const int laneHi = lane >> 4;                        // 0 / 1
    const int aOff = laneHi * 8;    // A: K split {0..7,16..23} / {8..15,24..31}
    const int bOff = laneHi * 8;    // B: K split {0..7,16..23} / {8..15,24..31}

    // ---- staging assignments (hoisted) ----
    const int sr = tid >> 1;                 // 0..127 (one LDS row per 2 threads)
    const int ss = (tid & 1) * 32;           // 32-half half-row
    int stRow = rowStart + sr;
    int stRowC = (stRow < cnt) ? stRow : cnt - 1;
    const _Float16* aSrc = hsH + (size_t)rowTok[off + stRowC] * HID + ss;
    const int wrow = (sr < 64) ? (c0 + sr) : (INTER + c0 + (sr - 64));
    const float* bSrc = gup + (size_t)e * (2 * INTER) * HID + (size_t)wrow * HID + ss;
    const int sOff = sr * LDSTR + ss;

    v8f acc[2][2][2] = {};                   // [rowsub][colsub][gate/up]

    // ---- prologue: load + store chunk 0 ----
    {
        const _Float16* s = aSrc;
        uint4 a0 = *(const uint4*)(s);
        uint4 a1 = *(const uint4*)(s + 8);
        uint4 a2 = *(const uint4*)(s + 16);
        uint4 a3 = *(const uint4*)(s + 24);
        float4 bF[8];
        const float* t = bSrc;
#pragma unroll
        for (int i = 0; i < 8; ++i) bF[i] = *(const float4*)(t + 4 * i);
        _Float16* da = ldsA[0] + sOff;
        *(uint4*)(da)      = a0;
        *(uint4*)(da + 8)  = a1;
        *(uint4*)(da + 16) = a2;
        *(uint4*)(da + 24) = a3;
        _Float16* db = ldsB[0] + sOff;
        *(uint4*)(db)      = pack8(bF[0], bF[1]);
        *(uint4*)(db + 8)  = pack8(bF[2], bF[3]);
        *(uint4*)(db + 16) = pack8(bF[4], bF[5]);
        *(uint4*)(db + 24) = pack8(bF[6], bF[7]);
    }
    __syncthreads();

    for (int kc = 0; kc < HID; kc += 64) {
        const int cur = (kc >> 6) & 1;
        const bool more = (kc + 64) < HID;

        float4 bF[8];
        if (more) {
            // A loads first: the A-store below only waits for these (in-order VMEM),
            // then copies straight into the free buffer (cur^1 was drained last iter).
            const _Float16* s = aSrc + kc + 64;
            uint4 a0 = *(const uint4*)(s);
            uint4 a1 = *(const uint4*)(s + 8);
            uint4 a2 = *(const uint4*)(s + 16);
            uint4 a3 = *(const uint4*)(s + 24);
            // B loads issued now, consumed after the WMMAs.
            const float* t = bSrc + kc + 64;
            __builtin_prefetch(t + 64, 0, 0);            // warm L2 one chunk further
#pragma unroll
            for (int i = 0; i < 8; ++i) bF[i] = *(const float4*)(t + 4 * i);
            _Float16* da = ldsA[cur ^ 1] + sOff;
            *(uint4*)(da)      = a0;
            *(uint4*)(da + 8)  = a1;
            *(uint4*)(da + 16) = a2;
            *(uint4*)(da + 24) = a3;
        }

        // ---- compute: 2 k-steps of 32, 8 WMMA each (B loads in flight) ----
        const _Float16* lA = ldsA[cur];
        const _Float16* lB = ldsB[cur];
#pragma unroll
        for (int ks = 0; ks < 64; ks += 32) {
            HFrag a[2];
#pragma unroll
            for (int rs = 0; rs < 2; ++rs) {
                const _Float16* ar =
                    lA + (rowGroup * 32 + rs * 16 + laneM) * LDSTR + ks + aOff;
                a[rs].q[0] = *(const uint4*)(ar);
                a[rs].q[1] = *(const uint4*)(ar + 16);
            }
#pragma unroll
            for (int cs = 0; cs < 2; ++cs) {
#pragma unroll
                for (int gu = 0; gu < 2; ++gu) {
                    const _Float16* br =
                        lB + (gu * 64 + colGroup * 32 + cs * 16 + laneM) * LDSTR + ks + bOff;
                    HFrag b;
                    b.q[0] = *(const uint4*)(br);
                    b.q[1] = *(const uint4*)(br + 16);
#pragma unroll
                    for (int rs = 0; rs < 2; ++rs) {
                        acc[rs][cs][gu] = __builtin_amdgcn_wmma_f32_16x16x32_f16(
                            false, a[rs].v, false, b.v, (short)0,
                            acc[rs][cs][gu], false, false);
                    }
                }
            }
        }

        // ---- convert + store B into the other buffer ----
        if (more) {
            _Float16* db = ldsB[cur ^ 1] + sOff;
            *(uint4*)(db)      = pack8(bF[0], bF[1]);
            *(uint4*)(db + 8)  = pack8(bF[2], bF[3]);
            *(uint4*)(db + 16) = pack8(bF[4], bF[5]);
            *(uint4*)(db + 24) = pack8(bF[6], bF[7]);
        }
        __syncthreads();
    }

    // ---- epilogue: act = silu(gate) * up, f16 store at sorted position ----
#pragma unroll
    for (int rs = 0; rs < 2; ++rs) {
#pragma unroll
        for (int cs = 0; cs < 2; ++cs) {
            const int col = c0 + colGroup * 32 + cs * 16 + laneM;
#pragma unroll
            for (int r = 0; r < 8; ++r) {
                const int rowIdx = rowStart + rowGroup * 32 + rs * 16 + r + laneHi * 8;
                if (rowIdx < cnt) {
                    float g = acc[rs][cs][0][r];
                    float u = acc[rs][cs][1][r];
                    float sg = g / (1.0f + __expf(-g));   // silu
                    act[(size_t)(off + rowIdx) * INTER + col] = (_Float16)(sg * u);
                }
            }
        }
    }
}

// ---------------- phase B: act @ down[e]^T * w -> atomic add into out ----------------
// grid: (HID/64, TK/128, NUM_EXPERTS), block 256 = 8 waves.
// Block tile: 128 rows x 64 out cols. Wave: 32 rows x 32 cols. Same pipeline.
__global__ void __launch_bounds__(256, 2)
moe_down(const _Float16* __restrict__ act,              // [TK, I] f16, sorted rows
         const float* __restrict__ dwn,                 // [E, H, I] f32
         const int* __restrict__ counts,
         const int* __restrict__ offsets,
         const int* __restrict__ rowTok,
         const float* __restrict__ rowWgt,
         float* __restrict__ out)                       // [T, H] f32
{
    __shared__ __align__(16) _Float16 ldsA[2][128 * LDSTR];   // 2 x 18 KB
    __shared__ __align__(16) _Float16 ldsB[2][64 * LDSTR];    // 2 x 9 KB

    const int e = blockIdx.z;
    const int cnt = counts[e];
    const int rowStart = blockIdx.y * 128;
    if (rowStart >= cnt) return;
    const int off = offsets[e];
    const int c0 = blockIdx.x * 64;                      // output column base

    const int tid = threadIdx.x;
    const int lane = tid & 31;
    const int wave = tid >> 5;
    const int rowGroup = wave >> 1;
    const int colGroup = wave & 1;
    const int laneM = lane & 15;
    const int laneHi = lane >> 4;
    const int aOff = laneHi * 8;
    const int bOff = laneHi * 8;

    // ---- staging assignments ----
    const int sr = tid >> 1;                 // A: 0..127
    const int ss = (tid & 1) * 32;
    int stRow = rowStart + sr;
    int stRowC = (stRow < cnt) ? stRow : cnt - 1;
    const _Float16* aSrc = act + (size_t)(off + stRowC) * INTER + ss;   // contiguous rows
    const int aOffL = sr * LDSTR + ss;

    const int br_ = tid >> 2;                // B: 0..63 weight rows, 4 threads/row
    const int bs_ = (tid & 3) * 16;          // 16 halves each
    const float* bSrc = dwn + (size_t)e * HID * INTER + (size_t)(c0 + br_) * INTER + bs_;
    const int bOffL = br_ * LDSTR + bs_;

    v8f acc[2][2] = {};                      // [rowsub][colsub]

    // ---- prologue ----
    {
        const _Float16* s = aSrc;
        uint4 a0 = *(const uint4*)(s);
        uint4 a1 = *(const uint4*)(s + 8);
        uint4 a2 = *(const uint4*)(s + 16);
        uint4 a3 = *(const uint4*)(s + 24);
        float4 bF[4];
        const float* t = bSrc;
#pragma unroll
        for (int i = 0; i < 4; ++i) bF[i] = *(const float4*)(t + 4 * i);
        _Float16* da = ldsA[0] + aOffL;
        *(uint4*)(da)      = a0;
        *(uint4*)(da + 8)  = a1;
        *(uint4*)(da + 16) = a2;
        *(uint4*)(da + 24) = a3;
        _Float16* db = ldsB[0] + bOffL;
        *(uint4*)(db)     = pack8(bF[0], bF[1]);
        *(uint4*)(db + 8) = pack8(bF[2], bF[3]);
    }
    __syncthreads();

    for (int kc = 0; kc < INTER; kc += 64) {
        const int cur = (kc >> 6) & 1;
        const bool more = (kc + 64) < INTER;

        float4 bF[4];
        if (more) {
            const _Float16* s = aSrc + kc + 64;
            uint4 a0 = *(const uint4*)(s);
            uint4 a1 = *(const uint4*)(s + 8);
            uint4 a2 = *(const uint4*)(s + 16);
            uint4 a3 = *(const uint4*)(s + 24);
            const float* t = bSrc + kc + 64;
            __builtin_prefetch(t + 64, 0, 0);
#pragma unroll
            for (int i = 0; i < 4; ++i) bF[i] = *(const float4*)(t + 4 * i);
            _Float16* da = ldsA[cur ^ 1] + aOffL;
            *(uint4*)(da)      = a0;
            *(uint4*)(da + 8)  = a1;
            *(uint4*)(da + 16) = a2;
            *(uint4*)(da + 24) = a3;
        }

        const _Float16* lA = ldsA[cur];
        const _Float16* lB = ldsB[cur];
#pragma unroll
        for (int ks = 0; ks < 64; ks += 32) {
            HFrag a[2];
#pragma unroll
            for (int rs = 0; rs < 2; ++rs) {
                const _Float16* ar =
                    lA + (rowGroup * 32 + rs * 16 + laneM) * LDSTR + ks + aOff;
                a[rs].q[0] = *(const uint4*)(ar);
                a[rs].q[1] = *(const uint4*)(ar + 16);
            }
#pragma unroll
            for (int cs = 0; cs < 2; ++cs) {
                const _Float16* brp =
                    lB + (colGroup * 32 + cs * 16 + laneM) * LDSTR + ks + bOff;
                HFrag b;
                b.q[0] = *(const uint4*)(brp);
                b.q[1] = *(const uint4*)(brp + 16);
#pragma unroll
                for (int rs = 0; rs < 2; ++rs) {
                    acc[rs][cs] = __builtin_amdgcn_wmma_f32_16x16x32_f16(
                        false, a[rs].v, false, b.v, (short)0, acc[rs][cs], false, false);
                }
            }
        }

        if (more) {
            _Float16* db = ldsB[cur ^ 1] + bOffL;
            *(uint4*)(db)     = pack8(bF[0], bF[1]);
            *(uint4*)(db + 8) = pack8(bF[2], bF[3]);
        }
        __syncthreads();
    }

    // ---- epilogue: scale by routing weight, atomic add into out[token] ----
    int   toks[2][8];
    float wgts[2][8];
#pragma unroll
    for (int rs = 0; rs < 2; ++rs) {
#pragma unroll
        for (int r = 0; r < 8; ++r) {
            const int rowIdx = rowStart + rowGroup * 32 + rs * 16 + r + laneHi * 8;
            const int rc = (rowIdx < cnt) ? rowIdx : cnt - 1;
            toks[rs][r] = rowTok[off + rc];
            wgts[rs][r] = rowWgt[off + rc];
        }
    }
#pragma unroll
    for (int rs = 0; rs < 2; ++rs) {
#pragma unroll
        for (int cs = 0; cs < 2; ++cs) {
            const int col = c0 + colGroup * 32 + cs * 16 + laneM;
#pragma unroll
            for (int r = 0; r < 8; ++r) {
                const int rowIdx = rowStart + rowGroup * 32 + rs * 16 + r + laneHi * 8;
                if (rowIdx < cnt) {
                    atomicAdd(out + (size_t)toks[rs][r] * HID + col,
                              wgts[rs][r] * acc[rs][cs][r]);
                }
            }
        }
    }
}

// ---------------- launch ----------------
extern "C" void kernel_launch(void* const* d_in, const int* in_sizes, int n_in,
                              void* d_out, int out_size, void* d_ws, size_t ws_size,
                              hipStream_t stream) {
    const float* hs       = (const float*)d_in[0];      // [T, H]
    const int*   topk_idx = (const int*)d_in[1];        // [T, K]
    const float* topk_wgt = (const float*)d_in[2];      // [T, K]
    const float* gup      = (const float*)d_in[3];      // [E, 2I, H]
    const float* dwn      = (const float*)d_in[4];      // [E, H, I]
    float* out            = (float*)d_out;              // [T, H]

    // workspace layout
    char* ws = (char*)d_ws;
    int*   counts  = (int*)(ws + 0);
    int*   cursors = (int*)(ws + 64);
    int*   offsets = (int*)(ws + 128);
    int*   rowTok  = (int*)(ws + 1024);
    float* rowWgt  = (float*)(ws + 1024 + sizeof(int) * TK);
    size_t hsOff   = 131072;                                        // 128 KB aligned
    _Float16* hsH  = (_Float16*)(ws + hsOff);                       // 16 MB
    _Float16* actB = (_Float16*)(ws + hsOff + (size_t)2 * NUM_TOKENS * HID);  // 32 MB

    // routing
    moe_route_zero<<<1, 32, 0, stream>>>(counts, cursors);
    moe_route_count<<<(TK + 255) / 256, 256, 0, stream>>>(topk_idx, counts);
    moe_route_scan<<<1, 32, 0, stream>>>(counts, offsets);
    moe_route_fill<<<(TK + 255) / 256, 256, 0, stream>>>(topk_idx, topk_wgt,
                                                         offsets, cursors, rowTok, rowWgt);
    // convert hidden to f16; zero output
    moe_cvt_hs<<<(NUM_TOKENS * HID / 4 + 255) / 256, 256, 0, stream>>>(hs, hsH);
    moe_zero_out<<<(NUM_TOKENS * HID / 4 + 255) / 256, 256, 0, stream>>>(out);

    // grouped GEMM phase A: gate_up + SwiGLU (double-buffered LDS)
    dim3 gridA(INTER / 64, TK / 128, NUM_EXPERTS);
    moe_gateup_swiglu<<<gridA, 256, 0, stream>>>(hsH, gup, counts, offsets, rowTok, actB);

    // grouped GEMM phase B: down projection + weighted combine (double-buffered LDS)
    dim3 gridB(HID / 64, TK / 128, NUM_EXPERTS);
    moe_down<<<gridB, 256, 0, stream>>>(actB, dwn, counts, offsets, rowTok, rowWgt, out);
}
